// xLSTMPositionNet_3891240370985
// MI455X (gfx1250) — matrix operand order repaired and checked by reading, not compiled, not run on hardware
//
#include <hip/hip_runtime.h>
#include <hip/hip_bf16.h>
#include <math.h>

typedef __attribute__((ext_vector_type(16))) __bf16 bf16x16;
typedef __attribute__((ext_vector_type(8)))  float  f32x8;

#define NB  4096
#define NTS 512
#define NF  64
#define NH  32
#define NE  8
#define PFD 8   // prefetch distance in time steps

__device__ __forceinline__ float fast_rcp(float x) {
  return __builtin_amdgcn_rcpf(x);      // v_rcp_f32, ~1 ulp
}

__device__ __forceinline__ float fast_tanh(float x) {
  float xc = fminf(fmaxf(x, -15.0f), 15.0f);
  float e  = __expf(2.0f * xc);
  return (e - 1.0f) * fast_rcp(e + 1.0f);
}

__global__ __launch_bounds__(128, 1)
void xlstm_fused(const float* __restrict__ x,
                 const int*   __restrict__ ticker_id,
                 const float* __restrict__ wf_w, const float* __restrict__ wf_b,
                 const float* __restrict__ wi_w, const float* __restrict__ wi_b,
                 const float* __restrict__ wo_w, const float* __restrict__ wo_b,
                 const float* __restrict__ wz_w, const float* __restrict__ wz_b,
                 const float* __restrict__ rf_w, const float* __restrict__ ri_w,
                 const float* __restrict__ ro_w, const float* __restrict__ rz_w,
                 const float* __restrict__ emb,
                 const float* __restrict__ fc_w, const float* __restrict__ fc_b,
                 float* __restrict__ out)
{
  __shared__ float   sPre[4][2][32][8];   // [gate][Ntile][lane][i]  8 KB
  __shared__ __bf16  sH[16][34];          // h in bf16, row stride 68B (17 banks)
  __shared__ float   sHf[16][33];         // final-step h in f32

  const int tid  = threadIdx.x;
  const int wave = tid >> 5;              // gate id: 0=f 1=i 2=o 3=z
  const int lane = tid & 31;
  const int half = lane >> 4;             // 0 or 1
  const int lm   = lane & 15;
  const int b0   = blockIdx.x * 16;

  const float *wPtr, *bPtr, *rPtr;
  if      (wave == 0) { wPtr = wf_w; bPtr = wf_b; rPtr = rf_w; }
  else if (wave == 1) { wPtr = wi_w; bPtr = wi_b; rPtr = ri_w; }
  else if (wave == 2) { wPtr = wo_w; bPtr = wo_b; rPtr = ro_w; }
  else                { wPtr = wz_w; bPtr = wz_b; rPtr = rz_w; }

  // ---- load weights once into WMMA B-layout registers ----
  // B (K x 16): lane j -> column n = tile*16 + (j&15); element i -> K = (i<8 ? half*8+i : 16+half*8+i-8)
  bf16x16 WB[2][2];   // [Ntile][Kchunk]  for W^T  (K = 64)
  bf16x16 RB[2];      // [Ntile]          for R^T  (K = 32)
  #pragma unroll
  for (int tile = 0; tile < 2; ++tile) {
    const int n = tile * 16 + lm;
    #pragma unroll
    for (int chunk = 0; chunk < 2; ++chunk) {
      const float* src = wPtr + n * NF + chunk * 32 + half * 8;
      #pragma unroll
      for (int i = 0; i < 16; ++i) WB[tile][chunk][i] = (__bf16)src[i < 8 ? i : i + 8];
    }
    const float* rs = rPtr + n * NH + half * 8;
    #pragma unroll
    for (int i = 0; i < 16; ++i) RB[tile][i] = (__bf16)rs[i < 8 ? i : i + 8];
  }
  const float biasv[2] = { bPtr[lm], bPtr[16 + lm] };

  // elementwise partition for this wave: one N-tile, 4 accumulator rows
  const int tile_e = wave >> 1;           // N tile handled in elementwise stage
  const int ib     = (wave & 1) * 4;      // base accumulator index (0 or 4)

  // zero initial hidden state in LDS
  for (int idx = tid; idx < 16 * 34; idx += 128)
    ((__bf16*)sH)[idx] = (__bf16)0.0f;

  // c, n state for this wave's 4 elements (disjoint partition across waves)
  float cS[4] = {0.f, 0.f, 0.f, 0.f};
  float nS[4] = {0.f, 0.f, 0.f, 0.f};
  float hv[4] = {0.f, 0.f, 0.f, 0.f};

  __syncthreads();

  const float* xrow = x + (size_t)(b0 + lm) * NTS * NF;

  for (int t = 0; t < NTS; ++t) {
    // ---- A operand: x_t tile, 16-bit A layout ----
    // lane j -> row m = j&15; element i -> K = chunk*32 + (i<8 ? half*8+i : 16+half*8+i-8)
    const float* xp = xrow + t * NF;
    bf16x16 XA[2];
    #pragma unroll
    for (int chunk = 0; chunk < 2; ++chunk) {
      const float* p = xp + chunk * 32 + half * 8;
      float4 a0 = *(const float4*)(p);
      float4 a1 = *(const float4*)(p + 4);
      float4 c0 = *(const float4*)(p + 16);
      float4 c1 = *(const float4*)(p + 20);
      XA[chunk][0]  = (__bf16)a0.x; XA[chunk][1]  = (__bf16)a0.y;
      XA[chunk][2]  = (__bf16)a0.z; XA[chunk][3]  = (__bf16)a0.w;
      XA[chunk][4]  = (__bf16)a1.x; XA[chunk][5]  = (__bf16)a1.y;
      XA[chunk][6]  = (__bf16)a1.z; XA[chunk][7]  = (__bf16)a1.w;
      XA[chunk][8]  = (__bf16)c0.x; XA[chunk][9]  = (__bf16)c0.y;
      XA[chunk][10] = (__bf16)c0.z; XA[chunk][11] = (__bf16)c0.w;
      XA[chunk][12] = (__bf16)c1.x; XA[chunk][13] = (__bf16)c1.y;
      XA[chunk][14] = (__bf16)c1.z; XA[chunk][15] = (__bf16)c1.w;
      // hide HBM latency for the strided per-row stream
      if (t + PFD < NTS) __builtin_prefetch(p + PFD * NF, 0, 0);
    }
    // ---- A operand: h_{t-1} from LDS ----
    bf16x16 HA;
    {
      const int kb = half * 8;
      #pragma unroll
      for (int i = 0; i < 8; ++i) {
        HA[i]     = sH[lm][kb + i];
        HA[8 + i] = sH[lm][16 + kb + i];
      }
    }

    // ---- pre-activation = bias + x W^T + h R^T ----
    // Two independent accumulator chains, interleaved so each WMMA has a
    // non-dependent WMMA between it and its consumer (hides XDL latency).
    f32x8 acc0, acc1;
    #pragma unroll
    for (int i = 0; i < 8; ++i) { acc0[i] = biasv[0]; acc1[i] = biasv[1]; }
    acc0 = __builtin_amdgcn_wmma_f32_16x16x32_bf16(false, XA[0], false, WB[0][0], (short)0, acc0, false, false);
    acc1 = __builtin_amdgcn_wmma_f32_16x16x32_bf16(false, XA[0], false, WB[1][0], (short)0, acc1, false, false);
    acc0 = __builtin_amdgcn_wmma_f32_16x16x32_bf16(false, XA[1], false, WB[0][1], (short)0, acc0, false, false);
    acc1 = __builtin_amdgcn_wmma_f32_16x16x32_bf16(false, XA[1], false, WB[1][1], (short)0, acc1, false, false);
    acc0 = __builtin_amdgcn_wmma_f32_16x16x32_bf16(false, HA,    false, RB[0],    (short)0, acc0, false, false);
    acc1 = __builtin_amdgcn_wmma_f32_16x16x32_bf16(false, HA,    false, RB[1],    (short)0, acc1, false, false);
    #pragma unroll
    for (int i = 0; i < 8; ++i) {
      sPre[wave][0][lane][i] = acc0[i];
      sPre[wave][1][lane][i] = acc1[i];
    }
    __syncthreads();

    // ---- sLSTM elementwise update: partitioned, 4 elements per lane per wave ----
    #pragma unroll
    for (int ii = 0; ii < 4; ++ii) {
      const int i = ib + ii;
      float pf = sPre[0][tile_e][lane][i];
      float pi = sPre[1][tile_e][lane][i];
      float po = sPre[2][tile_e][lane][i];
      float pz = sPre[3][tile_e][lane][i];
      float o  = fast_rcp(1.0f + __expf(-po));
      float z  = fast_tanh(pz);
      float fh = __expf(fminf(pf, 10.0f));
      float ih = __expf(fminf(pi, 10.0f));
      float inv = fast_rcp(fh + ih + 1e-8f);
      float fg = fh * inv;
      float ig = ih * inv;
      float c  = fg * cS[ii] + ig * z;
      float nn = fg * nS[ii] + ig;
      cS[ii] = c;
      nS[ii] = nn;
      float h = o * (c * fast_rcp(nn + 1e-8f));
      hv[ii] = h;
      // C/D layout: M = half*8 + i, N = tile_e*16 + (lane&15)
      sH[half * 8 + i][tile_e * 16 + lm] = (__bf16)h;
    }
    __syncthreads();
  }

  // publish final-step h in f32 for the head
  #pragma unroll
  for (int ii = 0; ii < 4; ++ii)
    sHf[half * 8 + ib + ii][tile_e * 16 + lm] = hv[ii];
  __syncthreads();

  // ---- head: tanh([h, emb[ticker]] . fc_w + fc_b) ----
  if (tid < 16) {
    const int b = b0 + tid;
    float acc = fc_b[0];
    #pragma unroll
    for (int n = 0; n < NH; ++n) acc += fc_w[n] * sHf[tid][n];
    const int tk = ticker_id[b];
    #pragma unroll
    for (int e = 0; e < NE; ++e) acc += fc_w[NH + e] * emb[tk * NE + e];
    out[b] = tanhf(acc);
  }
}

extern "C" void kernel_launch(void* const* d_in, const int* in_sizes, int n_in,
                              void* d_out, int out_size, void* d_ws, size_t ws_size,
                              hipStream_t stream) {
  (void)in_sizes; (void)n_in; (void)d_ws; (void)ws_size; (void)out_size;
  const float* x    = (const float*)d_in[0];
  const int*   tidp = (const int*)  d_in[1];
  const float* wf_w = (const float*)d_in[2];
  const float* wf_b = (const float*)d_in[3];
  const float* wi_w = (const float*)d_in[4];
  const float* wi_b = (const float*)d_in[5];
  const float* wo_w = (const float*)d_in[6];
  const float* wo_b = (const float*)d_in[7];
  const float* wz_w = (const float*)d_in[8];
  const float* wz_b = (const float*)d_in[9];
  const float* rf_w = (const float*)d_in[10];
  const float* ri_w = (const float*)d_in[11];
  const float* ro_w = (const float*)d_in[12];
  const float* rz_w = (const float*)d_in[13];
  const float* emb  = (const float*)d_in[14];
  const float* fc_w = (const float*)d_in[15];
  const float* fc_b = (const float*)d_in[16];
  float* out = (float*)d_out;

  dim3 grid(NB / 16), block(128);
  hipLaunchKernelGGL(xlstm_fused, grid, block, 0, stream,
                     x, tidp, wf_w, wf_b, wi_w, wi_b, wo_w, wo_b, wz_w, wz_b,
                     rf_w, ri_w, ro_w, rz_w, emb, fc_w, fc_b, out);
}
